// GNNPolicy_79817672229039
// MI455X (gfx1250) — compile-verified, matrix operand order checked by reading, not run on hardware
//
#include <hip/hip_runtime.h>
#include <hip/hip_bf16.h>
#include <math.h>

// Problem constants (from reference): N=128 nodes, E=4096 edges, D=130, H=64.
#define NN 128
#define EE 4096
#define DD 130
#define HH 64

typedef __attribute__((ext_vector_type(16))) __bf16 v16bf;
typedef __attribute__((ext_vector_type(8)))  float  v8f;

union Frag16 { v16bf v; unsigned short s[16]; unsigned int u[8]; };

// Native f32 -> bf16 (RNE); lowers to gfx1250's bf16 convert instead of a
// 3-op bit-trick sequence (v_bfe + v_add3 + v_mov_b16).
__device__ __forceinline__ unsigned short f2bf(float f) {
  __bf16 h = (__bf16)f;
  return __builtin_bit_cast(unsigned short, h);
}

// K offset of dword-slot v for the 16-bit A/B fragment layout (wave32):
// v<4 -> K = 2v (+8 for lanes 16..31); v>=4 -> K = 16 + 2(v-4) (+8 high half).
// Always even -> (K, K+1) pairs are contiguous.
__device__ __forceinline__ int kof(int v, int hi) {
  return (v < 4 ? v * 2 : 16 + (v - 4) * 2) + hi * 8;
}

__device__ __forceinline__ v8f zero8() {
  v8f z;
#pragma unroll
  for (int i = 0; i < 8; ++i) z[i] = 0.0f;
  return z;
}

// Z is stored j-pair interleaved: element (j,n,o) lives at
//   ZP(j,n,o) = (((j>>1)*128 + n)*64 + o)*2 + (j&1)
// so a B-fragment dword slot (j0 even) is ONE aligned u32 load.
__device__ __forceinline__ size_t zpair(int j, int n, int o) {
  return ((((size_t)(j >> 1)) * NN + n) * HH + o) * 2 + (j & 1);
}

// ---------------------------------------------------------------------------
// Z[j][n][o] = sum_k A[n][k] * Wb[j][k*64+o]   (bf16 out, fp32 accumulate)
// One block per j (256 thr = 8 waves; wave w owns rows m0=w*16, loops 4 o-tiles)
// KD is compile-time: full k-tiles are branch-free; tail uses clamp+select.
// ---------------------------------------------------------------------------
template <int KD>
__global__ void z_gemm_kernel(const float* __restrict__ A,    // [128 x KD]
                              const float* __restrict__ Wb,   // [J x KD*64]
                              unsigned short* __restrict__ Z) {
  constexpr int KFULL = KD / 32;
  constexpr int TAIL  = KD % 32;
  const int j    = blockIdx.x;
  const int wave = threadIdx.x >> 5;
  const int lane = threadIdx.x & 31;
  const int lo   = lane & 15;
  const int hi   = lane >> 4;
  const int m0   = wave * 16;
  const float* wrow = Wb + (size_t)j * (KD * HH);
  const float* arow = A + (size_t)(m0 + lo) * KD;

  v8f c[4];
#pragma unroll
  for (int ot = 0; ot < 4; ++ot) c[ot] = zero8();

  for (int kt = 0; kt < KFULL; ++kt) {
    const int kbase = kt * 32;
    if (kt + 1 < KFULL)
      __builtin_prefetch(wrow + (size_t)(kbase + 32) * HH, 0, 1);
    Frag16 a;
#pragma unroll
    for (int v = 0; v < 8; ++v) {
      const float2 p = *(const float2*)(arow + kbase + kof(v, hi));
      a.s[2 * v] = f2bf(p.x);  a.s[2 * v + 1] = f2bf(p.y);
    }
#pragma unroll
    for (int ot = 0; ot < 4; ++ot) {
      const int o = ot * 16 + lo;
      Frag16 b;
#pragma unroll
      for (int v = 0; v < 8; ++v) {
        const int k0 = kbase + kof(v, hi);
        b.s[2 * v]     = f2bf(wrow[(size_t)k0 * HH + o]);
        b.s[2 * v + 1] = f2bf(wrow[(size_t)(k0 + 1) * HH + o]);
      }
      c[ot] = __builtin_amdgcn_wmma_f32_16x16x32_bf16(false, a.v, false, b.v,
                                                      (short)0, c[ot], false, false);
    }
  }

  if (TAIL) {   // clamp addresses, select values to zero OOB -> no exec branches
    const int kbase = KFULL * 32;
    Frag16 a;
#pragma unroll
    for (int v = 0; v < 8; ++v) {
      const int k0 = kbase + kof(v, hi);
      const int kc = min(k0, KD - 2);
      const float2 p = *(const float2*)(arow + kc);
      a.s[2 * v]     = (k0     < KD) ? f2bf(p.x) : (unsigned short)0;
      a.s[2 * v + 1] = (k0 + 1 < KD) ? f2bf(p.y) : (unsigned short)0;
    }
#pragma unroll
    for (int ot = 0; ot < 4; ++ot) {
      const int o = ot * 16 + lo;
      Frag16 b;
#pragma unroll
      for (int v = 0; v < 8; ++v) {
        const int k0  = kbase + kof(v, hi);
        const int kc0 = min(k0, KD - 1);
        const int kc1 = min(k0 + 1, KD - 1);
        const float w0 = wrow[(size_t)kc0 * HH + o];
        const float w1 = wrow[(size_t)kc1 * HH + o];
        b.s[2 * v]     = (k0     < KD) ? f2bf(w0) : (unsigned short)0;
        b.s[2 * v + 1] = (k0 + 1 < KD) ? f2bf(w1) : (unsigned short)0;
      }
      c[ot] = __builtin_amdgcn_wmma_f32_16x16x32_bf16(false, a.v, false, b.v,
                                                      (short)0, c[ot], false, false);
    }
  }

#pragma unroll
  for (int ot = 0; ot < 4; ++ot) {
    const int o = ot * 16 + lo;
#pragma unroll
    for (int v = 0; v < 8; ++v) {
      const int m = m0 + v + hi * 8;           // C layout: lanes 0-15 M=v, 16-31 M=v+8
      Z[zpair(j, m, o)] = f2bf(c[ot][v]);
    }
  }
}

// ---------------------------------------------------------------------------
// Per-edge message + scatter-mean accumulate.
// msg[e,o] = sum_j relu(a_e*Wa[j]+ba[j]) * Z[src,j,o] + bterm[src,o]
// Block = (src s, 16-edge tile t); 128 thr = 4 waves, wave w -> o-tile w.
// Y regenerated in registers (never materialized: saves 136 MB of traffic).
// Branch-free: invalid rows are zeroed by a multiplicative lane mask.
// ---------------------------------------------------------------------------
template <int J>
__global__ void edge_msg_kernel(const int* __restrict__ edgeIds,
                                const int* __restrict__ srcStart,   // [129]
                                const int* __restrict__ edge_index, // [2*E]
                                const float* __restrict__ edge_attr,// [E]
                                const float* __restrict__ Wa,       // [J]
                                const float* __restrict__ ba,       // [J]
                                const unsigned short* __restrict__ Z,
                                const float* __restrict__ bterm,    // [128*64]
                                float* __restrict__ agg) {          // [128*64]
  const int sn   = blockIdx.x >> 4;
  const int t    = blockIdx.x & 15;
  const int base = srcStart[sn];
  const int deg  = srcStart[sn + 1] - base;
  if (t * 16 >= deg) return;

  const int wave = threadIdx.x >> 5;
  const int lane = threadIdx.x & 31;
  const int lo   = lane & 15;
  const int hi   = lane >> 4;
  const int o    = wave * 16 + lo;

  // This lane's A-fragment row = edge (t*16 + lo) of the group (clamped load).
  const int   rowIdx = t * 16 + lo;
  const float vmask  = (rowIdx < deg) ? 1.0f : 0.0f;
  const float aAttr  = edge_attr[edgeIds[base + min(rowIdx, deg - 1)]];

  // u32 index of the (j0, j0+1) bf16 pair for this lane's (src, o) column.
  const unsigned int* zcol =
      (const unsigned int*)(Z) + ((size_t)sn * HH + o);
  v8f c = zero8();
  constexpr int KSTEPS = J / 32;              // J is 8320 or 64 (both % 32 == 0)
  for (int kt = 0; kt < KSTEPS; ++kt) {
    Frag16 a, b;
#pragma unroll
    for (int v = 0; v < 8; ++v) {
      const int j0 = kt * 32 + kof(v, hi);
      const float2 w2 = *(const float2*)(Wa + j0);
      const float2 b2 = *(const float2*)(ba + j0);
      a.s[2 * v]     = f2bf(fmaxf(fmaf(aAttr, w2.x, b2.x), 0.0f) * vmask);
      a.s[2 * v + 1] = f2bf(fmaxf(fmaf(aAttr, w2.y, b2.y), 0.0f) * vmask);
      b.u[v] = zcol[((size_t)(j0 >> 1)) * (NN * HH)];   // one u32 = bf16 pair
    }
    c = __builtin_amdgcn_wmma_f32_16x16x32_bf16(false, a.v, false, b.v,
                                                (short)0, c, false, false);
  }

  const float bt = bterm[sn * HH + o];
#pragma unroll
  for (int v = 0; v < 8; ++v) {
    const int mIdx = t * 16 + v + hi * 8;     // C layout row
    if (mIdx < deg) {
      const int e   = edgeIds[base + mIdx];
      const int dst = edge_index[EE + e];
      atomicAdd(&agg[dst * HH + o], c[v] + bt);
    }
  }
}

// ---------------------------------------------------------------------------
// Small helper kernels
// ---------------------------------------------------------------------------
__global__ void zero_kernel(float* p, int n) {
  const int i = blockIdx.x * blockDim.x + threadIdx.x;
  if (i < n) p[i] = 0.0f;
}

__global__ void hist_kernel(const int* __restrict__ edge_index,
                            int* __restrict__ srcCount, float* __restrict__ cnt) {
  const int e = blockIdx.x * blockDim.x + threadIdx.x;
  if (e < EE) {
    atomicAdd(&srcCount[edge_index[e]], 1);
    atomicAdd(&cnt[edge_index[EE + e]], 1.0f);
  }
}

__global__ void scan_kernel(const int* __restrict__ srcCount, int* __restrict__ srcStart) {
  if (threadIdx.x == 0) {
    int acc = 0;
    for (int i = 0; i < NN; ++i) { srcStart[i] = acc; acc += srcCount[i]; }
    srcStart[NN] = acc;
  }
}

__global__ void scatter_kernel(const int* __restrict__ edge_index,
                               const int* __restrict__ srcStart,
                               int* __restrict__ srcCursor, int* __restrict__ edgeIds) {
  const int e = blockIdx.x * blockDim.x + threadIdx.x;
  if (e < EE) {
    const int s = edge_index[e];
    const int pos = srcStart[s] + atomicAdd(&srcCursor[s], 1);
    edgeIds[pos] = e;
  }
}

// bterm[n,o] = sum_k A[n,k] * bvec[k*64+o]  (the b1b/b2b contribution to messages)
__global__ void bterm_kernel(const float* __restrict__ A, const float* __restrict__ bvec,
                             float* __restrict__ out, int Kdim) {
  const int tid = blockIdx.x * blockDim.x + threadIdx.x;
  if (tid >= NN * HH) return;
  const int n = tid >> 6, o = tid & 63;
  float acc = 0.0f;
  for (int k = 0; k < Kdim; ++k) acc = fmaf(A[(size_t)n * Kdim + k], bvec[k * HH + o], acc);
  out[tid] = acc;
}

// h[n,o] = relu(agg/max(cnt,1) + Ain @ Wr + bc)
__global__ void finish_kernel(const float* __restrict__ agg, const float* __restrict__ cnt,
                              const float* __restrict__ Ain, const float* __restrict__ Wr,
                              const float* __restrict__ bc, float* __restrict__ h, int Kin) {
  const int tid = blockIdx.x * blockDim.x + threadIdx.x;
  if (tid >= NN * HH) return;
  const int n = tid >> 6, o = tid & 63;
  float r = agg[tid] / fmaxf(cnt[n], 1.0f);
  float root = 0.0f;
  for (int k = 0; k < Kin; ++k) root = fmaf(Ain[(size_t)n * Kin + k], Wr[k * HH + o], root);
  h[tid] = fmaxf(r + root + bc[o], 0.0f);
}

// logits -> adjacency-masked softmax -> packet masking (one block per row)
__global__ void head_kernel(const float* __restrict__ h2, const float* __restrict__ Wout,
                            const float* __restrict__ bout, const int* __restrict__ adjacency,
                            const float* __restrict__ new_packets, float* __restrict__ out) {
  __shared__ float sh[NN];
  const int n = blockIdx.x, m = threadIdx.x;
  float acc = bout[m];
  for (int k = 0; k < HH; ++k) acc = fmaf(h2[n * HH + k], Wout[k * NN + m], acc);
  const float lg = (adjacency[n * NN + m] != 0) ? acc : -__builtin_inff();
  sh[m] = lg; __syncthreads();
  for (int s = 64; s > 0; s >>= 1) { if (m < s) sh[m] = fmaxf(sh[m], sh[m + s]); __syncthreads(); }
  const float mx = sh[0]; __syncthreads();
  const float ex = __expf(lg - mx);
  sh[m] = ex; __syncthreads();
  for (int s = 64; s > 0; s >>= 1) { if (m < s) sh[m] += sh[m + s]; __syncthreads(); }
  const float prob = ex / sh[0];
  const float np = new_packets[n];
  out[n * NN + m] = (np != -1.0f) ? prob : ((m == n) ? 1.0f : 0.0f);
}

// ---------------------------------------------------------------------------
extern "C" void kernel_launch(void* const* d_in, const int* in_sizes, int n_in,
                              void* d_out, int out_size, void* d_ws, size_t ws_size,
                              hipStream_t stream) {
  (void)in_sizes; (void)n_in; (void)out_size; (void)ws_size;
  const float* x           = (const float*)d_in[0];
  const int*   edge_index  = (const int*)  d_in[1];
  const float* edge_attr   = (const float*)d_in[2];
  const int*   adjacency   = (const int*)  d_in[3];
  const float* new_packets = (const float*)d_in[4];
  const float* W1a = (const float*)d_in[5];
  const float* b1a = (const float*)d_in[6];
  const float* W1b = (const float*)d_in[7];
  const float* b1b = (const float*)d_in[8];
  const float* Wr1 = (const float*)d_in[9];
  const float* bc1 = (const float*)d_in[10];
  const float* W2a = (const float*)d_in[11];
  const float* b2a = (const float*)d_in[12];
  const float* W2b = (const float*)d_in[13];
  const float* b2b = (const float*)d_in[14];
  const float* Wr2 = (const float*)d_in[15];
  const float* bc2 = (const float*)d_in[16];
  const float* Wout = (const float*)d_in[17];
  const float* bout = (const float*)d_in[18];

  char* ws = (char*)d_ws;
  size_t off = 0;
  auto alloc = [&](size_t bytes) -> void* {
    void* p = ws + off;
    off += (bytes + 255) & ~(size_t)255;
    return p;
  };
  // contiguous zero-init region first: agg1, agg2, cnt, srcCount, srcCursor
  float* agg1      = (float*)alloc((size_t)NN * HH * 4);
  float* agg2      = (float*)alloc((size_t)NN * HH * 4);
  float* cnt       = (float*)alloc((size_t)NN * 4);
  int*   srcCount  = (int*)  alloc((size_t)NN * 4);
  int*   srcCursor = (int*)  alloc((size_t)NN * 4);
  const int zeroFloats = NN * HH + NN * HH + NN + NN + NN;   // 16768
  int*   srcStart  = (int*)  alloc((size_t)(NN + 1) * 4);
  int*   edgeIds   = (int*)  alloc((size_t)EE * 4);
  float* bterm1    = (float*)alloc((size_t)NN * HH * 4);
  float* bterm2    = (float*)alloc((size_t)NN * HH * 4);
  float* h1        = (float*)alloc((size_t)NN * HH * 4);
  float* h2        = (float*)alloc((size_t)NN * HH * 4);
  unsigned short* Z2 = (unsigned short*)alloc((size_t)HH * NN * HH * 2);        // 1 MB
  unsigned short* Z1 = (unsigned short*)alloc((size_t)DD * HH * NN * HH * 2);   // 136 MB

  // --- setup: zero accumulators, bucket edges by src, in-degree counts ---
  zero_kernel<<<(zeroFloats + 255) / 256, 256, 0, stream>>>(agg1, zeroFloats);
  hist_kernel<<<EE / 256, 256, 0, stream>>>(edge_index, srcCount, cnt);
  scan_kernel<<<1, 32, 0, stream>>>(srcCount, srcStart);
  scatter_kernel<<<EE / 256, 256, 0, stream>>>(edge_index, srcStart, srcCursor, edgeIds);

  // --- conv1 (J = D*H = 8320): Z1 GEMM (8.9 GMAC), edge contraction (2.2 GMAC) ---
  z_gemm_kernel<DD><<<DD * HH, 256, 0, stream>>>(x, W1b, Z1);
  bterm_kernel<<<(NN * HH + 255) / 256, 256, 0, stream>>>(x, b1b, bterm1, DD);
  edge_msg_kernel<DD * HH><<<NN * 16, 128, 0, stream>>>(edgeIds, srcStart, edge_index,
                                                        edge_attr, W1a, b1a, Z1, bterm1, agg1);
  finish_kernel<<<(NN * HH + 255) / 256, 256, 0, stream>>>(agg1, cnt, x, Wr1, bc1, h1, DD);

  // --- conv2 (J = H = 64): identical structure, tiny ---
  z_gemm_kernel<HH><<<HH, 256, 0, stream>>>(h1, W2b, Z2);
  bterm_kernel<<<(NN * HH + 255) / 256, 256, 0, stream>>>(h1, b2b, bterm2, HH);
  edge_msg_kernel<HH><<<NN * 16, 128, 0, stream>>>(edgeIds, srcStart, edge_index,
                                                   edge_attr, W2a, b2a, Z2, bterm2, agg2);
  finish_kernel<<<(NN * HH + 255) / 256, 256, 0, stream>>>(agg2, cnt, h1, Wr2, bc2, h2, HH);

  // --- head: logits + masked softmax + packet masking ---
  head_kernel<<<NN, NN, 0, stream>>>(h2, Wout, bout, adjacency, new_packets, (float*)d_out);
}
